// KANGuard_11493332484324
// MI455X (gfx1250) — compile-verified, matrix operand order checked by reading, not run on hardware
//
#include <hip/hip_runtime.h>
#include <hip/hip_bf16.h>
#include <math.h>

typedef __attribute__((ext_vector_type(16))) _Float16 v16h;
typedef __attribute__((ext_vector_type(8)))  _Float16 v8h;
typedef __attribute__((ext_vector_type(8)))  float    v8f;

#define IN_DIM 256
#define HID    128
#define ACT_NONE 0
#define ACT_RELU 1
#define ACT_GELU 2

__device__ __forceinline__ float gelu_f(float v) {
    // jax.nn.gelu default (approximate=True, tanh form)
    const float c = 0.7978845608028654f;
    float t = tanhf(c * (v + 0.044715f * v * v * v));
    return 0.5f * v * (1.0f + t);
}

// Repack W[kdim,ndim] f32 row-major -> f16 "fragment-native" layout:
// Wf[ ((ct*(kdim/32) + k32)*32 + lane)*16 + i ]  where
//   ct=n/16, l16=n%16, k32=k/32, koff=k%32,
//   half=(koff>>3)&1, i=(koff&7)+((koff>>4)<<3), lane=half*16+l16.
// Then a wave's B-fragment for (k32, ct) is one contiguous v16h per lane.
__global__ void pack_w_kernel(const float* __restrict__ W, _Float16* __restrict__ Wf,
                              int kdim, int ndim) {
    int idx = blockIdx.x * blockDim.x + threadIdx.x;
    if (idx >= kdim * ndim) return;
    int k = idx / ndim, n = idx - (idx / ndim) * ndim;
    int ct = n >> 4, l16 = n & 15;
    int k32 = k >> 5, koff = k & 31;
    int hi = koff >> 3;
    int half = hi & 1;
    int i = (koff & 7) + ((hi >> 1) << 3);
    int lane = half * 16 + l16;
    Wf[((((size_t)ct * (kdim >> 5) + k32) * 32 + lane) << 4) + i] = (_Float16)W[idx];
}

// C[n_rows, NDIM] = act(A[n_rows, KDIM] @ W + bias), W pre-packed f16 fragments.
// Block = 16 output rows, 8 waves; wave w owns 16x16 col tiles ct = w, w+8, ...
template <int KDIM, int NDIM, int ACT, bool HAS_BIAS>
__global__ void gemm_wmma_kernel(const float* __restrict__ A,
                                 const _Float16* __restrict__ Wf,
                                 const float* __restrict__ bias,
                                 float* __restrict__ C,
                                 int n_rows) {
    __shared__ __align__(32) _Float16 lds_a[16 * KDIM];

    const int tid  = threadIdx.x;
    const int wave = tid >> 5;
    const int lane = tid & 31;
    const int half = lane >> 4;   // 0: lanes 0-15, 1: lanes 16-31
    const int l16  = lane & 15;
    const int row0 = blockIdx.x * 16;

    // Stage 16 x KDIM strip of A into LDS as f16 (clamped loads, select-zero OOB)
#pragma unroll 4
    for (int idx = tid; idx < 16 * KDIM; idx += 256) {
        int r = idx / KDIM;            // compile-time divisor
        int k = idx - r * KDIM;
        int gr = row0 + r;
        int grc = (gr < n_rows) ? gr : (n_rows - 1);
        float v = A[(size_t)grc * KDIM + k];
        if (gr >= n_rows) v = 0.0f;
        lds_a[r * KDIM + k] = (_Float16)v;
    }
    __syncthreads();

    constexpr int K32    = KDIM >> 5;
    constexpr int NTILES = NDIM >> 4;
    const bool full_tile = (row0 + 16 <= n_rows);

#pragma unroll
    for (int ct = wave; ct < NTILES; ct += 8) {
        v8f acc = {};
        const int ncol = ct * 16 + l16;
        const _Float16* wbase = Wf + ((((size_t)ct * K32) * 32 + lane) << 4);
#pragma unroll
        for (int t = 0; t < K32; ++t) {
            const int kk = t * 32;
            // A fragment: two contiguous 8-half runs per lane (ds_load_b128 x2)
            const v8h alo = *(const v8h*)&lds_a[l16 * KDIM + kk + half * 8];
            const v8h ahi = *(const v8h*)&lds_a[l16 * KDIM + kk + 16 + half * 8];
            v16h a;
#pragma unroll
            for (int i = 0; i < 8; ++i) { a[i] = alo[i]; a[i + 8] = ahi[i]; }
            // B fragment: one contiguous 32B load per lane (global_load_b128 x2)
            const v16h b = *(const v16h*)(wbase + ((size_t)t << 9)); // t*32*16 halves
            acc = __builtin_amdgcn_wmma_f32_16x16x32_f16(
                false, a, false, b, (short)0, acc, false, false);
        }

        float bv = HAS_BIAS ? bias[ncol] : 0.0f;
        float vals[8];
#pragma unroll
        for (int v = 0; v < 8; ++v) {
            float val = acc[v] + bv;
            if (ACT == ACT_RELU)      val = fmaxf(val, 0.0f);
            else if (ACT == ACT_GELU) val = gelu_f(val);
            vals[v] = val;
        }
        if (full_tile) {
#pragma unroll
            for (int v = 0; v < 8; ++v)
                C[(size_t)(row0 + v + half * 8) * NDIM + ncol] = vals[v];
        } else {
#pragma unroll
            for (int v = 0; v < 8; ++v) {
                int grow = row0 + v + half * 8;
                if (grow < n_rows) C[(size_t)grow * NDIM + ncol] = vals[v];
            }
        }
    }
}

__global__ void zero_kernel(float* __restrict__ p, size_t n) {
    size_t i = (size_t)blockIdx.x * blockDim.x + threadIdx.x;
    size_t stride = (size_t)gridDim.x * blockDim.x;
    for (; i < n; i += stride) p[i] = 0.0f;
}

// Mean-aggregation scatter: acc[dst] += y[src], cnt[dst] += 1.
__global__ void scatter_add_kernel(const float* __restrict__ y,
                                   const int* __restrict__ src,
                                   const int* __restrict__ dst,
                                   float* __restrict__ acc,
                                   float* __restrict__ cnt,
                                   int E) {
    const int chunks = HID >> 2;   // 32
    long long idx = (long long)blockIdx.x * blockDim.x + threadIdx.x;
    long long total = (long long)E * chunks;
    if (idx >= total) return;
    int e = (int)(idx >> 5);
    int c = (int)(idx & 31);
    int s = src[e], d = dst[e];
    const float4 v = *(const float4*)(y + (size_t)s * HID + c * 4);
    float* ap = acc + (size_t)d * HID + c * 4;
    atomicAdd(ap + 0, v.x);
    atomicAdd(ap + 1, v.y);
    atomicAdd(ap + 2, v.z);
    atomicAdd(ap + 3, v.w);
    if (c == 0) atomicAdd(&cnt[d], 1.0f);
}

// out = relu(acc / max(cnt,1) + bias + hr)
__global__ void combine_relu_kernel(const float* __restrict__ acc,
                                    const float* __restrict__ cnt,
                                    const float* __restrict__ hr,
                                    const float* __restrict__ bias,
                                    float* __restrict__ out,
                                    int n) {
    size_t i = (size_t)blockIdx.x * blockDim.x + threadIdx.x;
    size_t total = (size_t)n * HID;
    if (i >= total) return;
    int row = (int)(i >> 7);
    int col = (int)(i & (HID - 1));
    float c = cnt[row];
    float v = acc[i] / fmaxf(c, 1.0f) + bias[col] + hr[i];
    out[i] = fmaxf(v, 0.0f);
}

// logits[row] = h[row,:] . Wc + bc   (one wave32 per row)
__global__ void classifier_kernel(const float* __restrict__ h,
                                  const float* __restrict__ Wc,
                                  const float* __restrict__ bc,
                                  float* __restrict__ out, int n) {
    int wid  = (int)(((size_t)blockIdx.x * blockDim.x + threadIdx.x) >> 5);
    int lane = threadIdx.x & 31;
    if (wid >= n) return;                    // uniform per wave
    const float* row = h + (size_t)wid * HID;
    float s = 0.0f;
#pragma unroll
    for (int k = lane; k < HID; k += 32) s += row[k] * Wc[k];
#pragma unroll
    for (int off = 16; off > 0; off >>= 1) s += __shfl_down(s, off);
    if (lane == 0) out[wid] = s + bc[0];
}

extern "C" void kernel_launch(void* const* d_in, const int* in_sizes, int n_in,
                              void* d_out, int out_size, void* d_ws, size_t ws_size,
                              hipStream_t stream) {
    const float* x    = (const float*)d_in[0];
    const int*   ei   = (const int*)  d_in[1];
    const float* W_l1 = (const float*)d_in[2];
    const float* b_l1 = (const float*)d_in[3];
    const float* W_r1 = (const float*)d_in[4];
    const float* W_l2 = (const float*)d_in[5];
    const float* b_l2 = (const float*)d_in[6];
    const float* W_r2 = (const float*)d_in[7];
    const float* W_k1 = (const float*)d_in[8];
    const float* b_k1 = (const float*)d_in[9];
    const float* W_k2 = (const float*)d_in[10];
    const float* b_k2 = (const float*)d_in[11];
    const float* W_c  = (const float*)d_in[12];
    const float* b_c  = (const float*)d_in[13];

    const int N = in_sizes[0] / IN_DIM;
    const int E = in_sizes[1] / 2;
    const int* src = ei;
    const int* dst = ei + E;

    float* ws = (float*)d_ws;
    const size_t sh = (size_t)N * HID;
    float* bufA = ws;            // N*HID (h1 / part of bufF)
    float* bufC = ws + sh;       // N*HID (agg accumulator / part of bufF)
    float* bufE = ws + 2 * sh;   // N*HID (h_r2 / part of bufF)
    float* bufF = bufA;          // N*4*HID KAN hidden (spans bufA..+4sh)
    float* bufB = ws + 4 * sh;   // N*HID (h_r1 / y_l2 / h2 / h4)
    float* cnt  = ws + 5 * sh;   // N floats
    // packed f16 weights (fragment-native) after cnt, 16-float aligned
    _Float16* wfbase = (_Float16*)(ws + ((5 * sh + (size_t)N + 15) & ~(size_t)15));
    _Float16* wf_l1 = wfbase;                 // 256*128
    _Float16* wf_r1 = wf_l1 + IN_DIM * HID;   // 256*128
    _Float16* wf_l2 = wf_r1 + IN_DIM * HID;   // 128*128
    _Float16* wf_r2 = wf_l2 + HID * HID;      // 128*128
    _Float16* wf_k1 = wf_r2 + HID * HID;      // 128*512
    _Float16* wf_k2 = wf_k1 + HID * 4 * HID;  // 512*128

    const dim3 blk(256);
    const int gm       = (N + 15) / 16;
    const int gzero128 = (int)((sh + 255) / 256);
    const int gzeroN   = (N + 255) / 256;
    const int gscat    = (int)(((long long)E * (HID / 4) + 255) / 256);
    const int gcomb    = (int)((sh + 255) / 256);
    const int gclass   = (N * 32 + 255) / 256;
    const int gpackL   = (IN_DIM * HID + 255) / 256;
    const int gpackH   = (HID * HID + 255) / 256;
    const int gpackK   = (HID * 4 * HID + 255) / 256;

    // ---- pack weights into fragment-native f16 (cheap, done every launch)
    pack_w_kernel<<<gpackL, blk, 0, stream>>>(W_l1, wf_l1, IN_DIM, HID);
    pack_w_kernel<<<gpackL, blk, 0, stream>>>(W_r1, wf_r1, IN_DIM, HID);
    pack_w_kernel<<<gpackH, blk, 0, stream>>>(W_l2, wf_l2, HID, HID);
    pack_w_kernel<<<gpackH, blk, 0, stream>>>(W_r2, wf_r2, HID, HID);
    pack_w_kernel<<<gpackK, blk, 0, stream>>>(W_k1, wf_k1, HID, 4 * HID);
    pack_w_kernel<<<gpackK, blk, 0, stream>>>(W_k2, wf_k2, 4 * HID, HID);

    // ---- SAGE layer 1 (project before aggregation: mean commutes with linear)
    gemm_wmma_kernel<IN_DIM, HID, ACT_NONE, false><<<gm, blk, 0, stream>>>(x, wf_l1, nullptr, bufA, N);
    gemm_wmma_kernel<IN_DIM, HID, ACT_NONE, false><<<gm, blk, 0, stream>>>(x, wf_r1, nullptr, bufB, N);
    zero_kernel<<<gzero128, blk, 0, stream>>>(bufC, sh);
    zero_kernel<<<gzeroN,   blk, 0, stream>>>(cnt, (size_t)N);
    scatter_add_kernel<<<gscat, blk, 0, stream>>>(bufA, src, dst, bufC, cnt, E);
    combine_relu_kernel<<<gcomb, blk, 0, stream>>>(bufC, cnt, bufB, b_l1, bufA, N); // h1 -> bufA

    // ---- SAGE layer 2
    gemm_wmma_kernel<HID, HID, ACT_NONE, false><<<gm, blk, 0, stream>>>(bufA, wf_l2, nullptr, bufB, N);
    gemm_wmma_kernel<HID, HID, ACT_NONE, false><<<gm, blk, 0, stream>>>(bufA, wf_r2, nullptr, bufE, N);
    zero_kernel<<<gzero128, blk, 0, stream>>>(bufC, sh);
    zero_kernel<<<gzeroN,   blk, 0, stream>>>(cnt, (size_t)N);
    scatter_add_kernel<<<gscat, blk, 0, stream>>>(bufB, src, dst, bufC, cnt, E);
    combine_relu_kernel<<<gcomb, blk, 0, stream>>>(bufC, cnt, bufE, b_l2, bufB, N); // h2 -> bufB

    // ---- KAN MLP
    gemm_wmma_kernel<HID, 4 * HID, ACT_GELU, true><<<gm, blk, 0, stream>>>(bufB, wf_k1, b_k1, bufF, N); // h3
    gemm_wmma_kernel<4 * HID, HID, ACT_GELU, true><<<gm, blk, 0, stream>>>(bufF, wf_k2, b_k2, bufB, N); // h4

    // ---- classifier
    classifier_kernel<<<gclass, blk, 0, stream>>>(bufB, W_c, b_c, (float*)d_out, N);
}
